// LowRankMultiheadAttention_53618371723763
// MI455X (gfx1250) — compile-verified, matrix operand order checked by reading, not run on hardware
//
#include <hip/hip_runtime.h>

typedef __attribute__((ext_vector_type(16))) _Float16 v16h;
typedef __attribute__((ext_vector_type(8)))  _Float16 v8h;
typedef __attribute__((ext_vector_type(8)))  float    v8f;
typedef __attribute__((ext_vector_type(4)))  unsigned int v4u;
typedef __attribute__((ext_vector_type(8)))  int      v8i;
typedef __attribute__((ext_vector_type(4)))  int      v4i;

#define IN_DIM 1024
#define EMB    1024
#define NH     16
#define HD     64
#define BATCH  4
#define TB     2048
#define T_KV   2049   // prompt + 2048
#define T_PAD  2080   // padded to 65 * 32
#define QLEN   801
#define QPAD   816    // padded to 51 * 16
#define QTILES (QPAD / 16)

// LDS row strides (halfs) after TDM padding: K 64+8, V/P 32+8
#define KLDS 72
#define VLDS 40

union frag16 { v16h v; v8h h[2]; };

// CDNA5 16-bit A/B fragment: lane L holds row (L%16); K sub-block base = (L/16)*8;
// halves [base..base+7] and [base+16..base+23] -> two b128 loads.
__device__ inline v16h load_frag(const _Float16* p) {
  frag16 f;
  f.h[0] = *reinterpret_cast<const v8h*>(p);
  f.h[1] = *reinterpret_cast<const v8h*>(p + 16);
  return f.v;
}

__device__ inline v8f wmma_f16(v16h a, v16h b, v8f c) {
  return __builtin_amdgcn_wmma_f32_16x16x32_f16(false, a, false, b, (short)0, c, false, false);
}

// ---- Tensor Data Mover: 2D tile load, 2-byte elements, optional LDS padding ----
// D# layout per CDNA5 ISA (groups 2/3 zero => 2D).
// This toolchain exposes the 6-arg builtin: (v4u, v8i, v4i, v4i, v8i, i32 cpol).
__device__ inline void tdm_load_2d(unsigned lds_addr, unsigned long long gaddr,
                                   unsigned tile_x, unsigned tile_y,
                                   unsigned tensor_x, unsigned tensor_y,
                                   unsigned stride_elems,
                                   unsigned pad_en, unsigned pad_int_code,
                                   unsigned pad_amt_code) {
  v4u g0;
  g0.x = 1u;                                         // count=1, user descriptor
  g0.y = lds_addr;                                   // LDS byte address
  g0.z = (unsigned)(gaddr & 0xffffffffu);            // global addr [31:0]
  g0.w = (unsigned)((gaddr >> 32) & 0x01ffffffu)     // global addr [56:32]
         | 0x80000000u;                              // type = 2 ("image")
  v8i g1;
  g1[0] = (int)((1u << 16)                           // data_size = 1 -> 2 bytes
                | (pad_en << 20)
                | (pad_int_code << 22)
                | (pad_amt_code << 25));
  g1[1] = (int)((tensor_x & 0xffffu) << 16);         // tensor_dim0[15:0]
  g1[2] = (int)(((tensor_x >> 16) & 0xffffu)
                | ((tensor_y & 0xffffu) << 16));     // dim0[31:16] | dim1[15:0]
  g1[3] = (int)(((tensor_y >> 16) & 0xffffu)
                | ((tile_x & 0xffffu) << 16));       // dim1[31:16] | tile_dim0
  g1[4] = (int)(tile_y & 0xffffu);                   // tile_dim1 (tile_dim2 = 0)
  g1[5] = (int)stride_elems;                         // tensor_dim0_stride[31:0]
  g1[6] = 0;
  g1[7] = 0;
  v4i z4 = {0, 0, 0, 0};
  v8i z8 = {0, 0, 0, 0, 0, 0, 0, 0};
  __builtin_amdgcn_tensor_load_to_lds(g0, g1, z4, z4, z8, 0);
}

// ---------------- staging kernels ----------------

__global__ void build_kv(const float* __restrict__ xb, const float* __restrict__ prompt,
                         const float* __restrict__ pe, _Float16* __restrict__ kv) {
  size_t i = (size_t)blockIdx.x * 256 + threadIdx.x;
  if (i >= (size_t)BATCH * T_PAD * IN_DIM) return;
  int k = (int)(i % IN_DIM);
  int r = (int)((i / IN_DIM) % T_PAD);
  int b = (int)(i / ((size_t)IN_DIM * T_PAD));
  float v = 0.f;
  if (r == 0)       v = prompt[k];
  else if (r <= TB) v = xb[((size_t)b * TB + (r - 1)) * IN_DIM + k] +
                        pe[(size_t)(r - 1) * IN_DIM + k];
  kv[i] = (_Float16)v;
}

__global__ void cvt_f16(const float* __restrict__ s, _Float16* __restrict__ d, int n) {
  int i = blockIdx.x * 256 + threadIdx.x;
  if (i < n) d[i] = (_Float16)s[i];
}

__global__ void build_posA(const float* __restrict__ pe, const int* __restrict__ fb,
                           const int* __restrict__ fa, _Float16* __restrict__ pa) {
  int i = blockIdx.x * 256 + threadIdx.x;
  if (i >= QPAD * IN_DIM) return;
  int k = i % IN_DIM, row = i / IN_DIM;
  float v = 0.f;
  if (row < QLEN) {
    int src = (row * fb[0]) / fa[0];          // arange(801)*50 // 801
    v = pe[(size_t)src * IN_DIM + k];
  }
  pa[i] = (_Float16)v;
}

__global__ void build_q(const float* __restrict__ dbg, const float* __restrict__ posP,
                        _Float16* __restrict__ q16) {
  size_t i = (size_t)blockIdx.x * 256 + threadIdx.x;
  if (i >= (size_t)BATCH * NH * QPAD * HD) return;
  int d  = (int)(i % HD);
  int qq = (int)((i / HD) % QPAD);
  int h  = (int)((i / ((size_t)HD * QPAD)) % NH);
  int b  = (int)(i / ((size_t)HD * QPAD * NH));
  float v = 0.f;
  if (qq < QLEN)
    v = dbg[(((size_t)(b * NH + h)) * QLEN + qq) * HD + d] +
        posP[(size_t)qq * EMB + h * HD + d];
  q16[i] = (_Float16)v;
}

// ---------------- WMMA GEMM: C(M,N) = A(MxK) * W(NxK)^T ----------------
// Each wave: 2 M-tiles x 4 N-tiles (32x64), 8 wmmas per 32-K step.
// mode 0: f16 row-major out (K-proj); mode 1: f16 transposed Vt[b][n][m];
// mode 2: f32 row-major out (pos-proj).
__global__ __launch_bounds__(256)
void gemm16(const _Float16* __restrict__ Abase, const _Float16* __restrict__ W,
            void* __restrict__ outp, int Mtiles, int mode) {
  const int lane = threadIdx.x & 31;
  const int w    = threadIdx.x >> 5;
  const int mtA  = blockIdx.y * 16 + w;       // first M tile
  if (mtA >= Mtiles) return;                  // wave-uniform
  const int mtB   = mtA + 8;                  // second M tile
  const bool hasB = (mtB < Mtiles);
  const int mtBl  = hasB ? mtB : mtA;         // clamped load base
  const int bz = blockIdx.z;
  const int n0 = blockIdx.x * 64;
  const int lr = lane & 15;
  const int k8 = (lane >> 4) << 3;

  const _Float16* A     = Abase + (mode < 2 ? (size_t)bz * T_PAD * IN_DIM : 0);
  const _Float16* arowA = A + (size_t)(mtA * 16 + lr) * IN_DIM + k8;
  const _Float16* arowB = A + (size_t)(mtBl * 16 + lr) * IN_DIM + k8;
  const _Float16* wrow  = W + (size_t)(n0 + lr) * IN_DIM + k8;

  v8f acc[2][4] = {};
  for (int kk = 0; kk < IN_DIM; kk += 32) {
    // all 6 fragments into distinct registers first -> back-to-back wmmas
    v16h a0 = load_frag(arowA + kk);
    v16h a1 = load_frag(arowB + kk);
    v16h b0 = load_frag(wrow + kk);
    v16h b1 = load_frag(wrow + (size_t)16 * IN_DIM + kk);
    v16h b2 = load_frag(wrow + (size_t)32 * IN_DIM + kk);
    v16h b3 = load_frag(wrow + (size_t)48 * IN_DIM + kk);
    acc[0][0] = wmma_f16(a0, b0, acc[0][0]);
    acc[0][1] = wmma_f16(a0, b1, acc[0][1]);
    acc[0][2] = wmma_f16(a0, b2, acc[0][2]);
    acc[0][3] = wmma_f16(a0, b3, acc[0][3]);
    acc[1][0] = wmma_f16(a1, b0, acc[1][0]);
    acc[1][1] = wmma_f16(a1, b1, acc[1][1]);
    acc[1][2] = wmma_f16(a1, b2, acc[1][2]);
    acc[1][3] = wmma_f16(a1, b3, acc[1][3]);
  }

  // C layout: lane holds rows (lane/16)*8 + r, col = tile*16 + lane%16
#pragma unroll
  for (int t = 0; t < 2; ++t) {
    if (t == 1 && !hasB) break;
    const int mt = t ? mtB : mtA;
    if (mode == 0) {
      _Float16* o = (_Float16*)outp;
#pragma unroll
      for (int j = 0; j < 4; ++j)
#pragma unroll
        for (int r = 0; r < 8; ++r) {
          int m = mt * 16 + k8 + r;
          o[((size_t)bz * T_PAD + m) * EMB + n0 + j * 16 + lr] = (_Float16)acc[t][j][r];
        }
    } else if (mode == 1) {
      _Float16* o = (_Float16*)outp;
#pragma unroll
      for (int j = 0; j < 4; ++j)
#pragma unroll
        for (int r = 0; r < 8; ++r) {
          int m = mt * 16 + k8 + r;
          o[(size_t)bz * EMB * T_PAD + (size_t)(n0 + j * 16 + lr) * T_PAD + m] =
              (_Float16)acc[t][j][r];
        }
    } else {
      float* o = (float*)outp;
#pragma unroll
      for (int j = 0; j < 4; ++j)
#pragma unroll
        for (int r = 0; r < 8; ++r) {
          int m = mt * 16 + k8 + r;
          o[(size_t)m * EMB + n0 + j * 16 + lr] = acc[t][j][r];
        }
    }
  }
}

// ---------------- flash attention with TDM-staged K/V tiles ----------------
// Block = 8 waves, one (b,h); all waves share the K/V tile -> wave 0 DMAs it
// into double-buffered LDS via TENSOR_LOAD_TO_LDS, s_wait_tensorcnt + barrier.
__global__ __launch_bounds__(256)
void attn(const _Float16* __restrict__ Qf, const _Float16* __restrict__ Kf,
          const _Float16* __restrict__ Vt, const float* __restrict__ gates,
          float* __restrict__ out) {
  __shared__ _Float16 kbuf[2][32][KLDS];   // 32 keys x 64 d (+8 pad)
  __shared__ _Float16 vbuf[2][64][VLDS];   // 64 d x 32 keys (+8 pad)
  __shared__ _Float16 pbuf[8][16][VLDS];   // per-wave P transpose tile

  const int lane = threadIdx.x & 31;
  const int w    = threadIdx.x >> 5;
  const int qt   = blockIdx.x * 8 + w;          // q tile (16 rows)
  const bool active = (qt < QTILES);            // wave-uniform
  const int h = blockIdx.y, b = blockIdx.z;
  const int lr = lane & 15;
  const int k8 = (lane >> 4) << 3;

  // Q fragments (A operand), hd=64 -> 2 k-steps of 32
  v16h q0 = {}, q1 = {};
  if (active) {
    const _Float16* qb = Qf + ((size_t)((b * NH + h) * QPAD) + qt * 16 + lr) * HD + k8;
    q0 = load_frag(qb);
    q1 = load_frag(qb + 32);
  }

  const unsigned long long kgbase =
      (unsigned long long)(uintptr_t)(Kf + ((size_t)b * T_PAD) * EMB + h * HD);
  const unsigned long long vgbase =
      (unsigned long long)(uintptr_t)(Vt + (size_t)b * EMB * T_PAD + (size_t)(h * HD) * T_PAD);

  float rm[8], rl[8];
  v8f o[4] = {};
#pragma unroll
  for (int r = 0; r < 8; ++r) { rm[r] = -1e30f; rl[r] = 0.f; }

  // prologue: DMA first K/V tile into buffer 0
  if (w == 0) {
    tdm_load_2d((unsigned)(uintptr_t)&kbuf[0][0][0], kgbase, 64, 32, 64, 32,
                EMB, 1u, 4u, 3u);                 // pad every 32 DW by 4 DW
    tdm_load_2d((unsigned)(uintptr_t)&vbuf[0][0][0], vgbase, 32, 64, 32, 64,
                T_PAD, 1u, 3u, 3u);               // pad every 16 DW by 4 DW
  }

  for (int t0 = 0; t0 < T_PAD; t0 += 32) {
    const int cur = (t0 >> 5) & 1;
    if (w == 0) __builtin_amdgcn_s_wait_tensorcnt((short)0);
    __syncthreads();                              // publish buf[cur]; prev reads done
    if (w == 0 && (t0 + 32) < T_PAD) {            // overlap next DMA with compute
      const int nxt = cur ^ 1;
      tdm_load_2d((unsigned)(uintptr_t)&kbuf[nxt][0][0],
                  kgbase + (size_t)(t0 + 32) * EMB * 2, 64, 32, 64, 32, EMB,
                  1u, 4u, 3u);
      tdm_load_2d((unsigned)(uintptr_t)&vbuf[nxt][0][0],
                  vgbase + (size_t)(t0 + 32) * 2, 32, 64, 32, 64, T_PAD,
                  1u, 3u, 3u);
    }
    if (!active) continue;

    // scores: 2 key sub-tiles x 2 k-steps; K fragments from LDS (conflict-free)
    v16h ka0 = load_frag(&kbuf[cur][lr][k8]);
    v16h ka1 = load_frag(&kbuf[cur][lr][32 + k8]);
    v16h kb0 = load_frag(&kbuf[cur][16 + lr][k8]);
    v16h kb1 = load_frag(&kbuf[cur][16 + lr][32 + k8]);
    v8f s0 = {}, s1 = {};
    s0 = wmma_f16(q0, ka0, s0);
    s0 = wmma_f16(q1, ka1, s0);
    s1 = wmma_f16(q0, kb0, s1);
    s1 = wmma_f16(q1, kb1, s1);

    const bool bad0 = (t0 + lr)      >= T_KV;     // mask padded keys
    const bool bad1 = (t0 + 16 + lr) >= T_KV;

    float p0[8], p1[8];
#pragma unroll
    for (int r = 0; r < 8; ++r) {
      float a = bad0 ? -1e30f : s0[r] * 0.125f;   // 1/sqrt(64)
      float c = bad1 ? -1e30f : s1[r] * 0.125f;
      p0[r] = a; p1[r] = c;
      float m = fmaxf(a, c);
      m = fmaxf(m, __shfl_xor(m, 1, 32));
      m = fmaxf(m, __shfl_xor(m, 2, 32));
      m = fmaxf(m, __shfl_xor(m, 4, 32));
      m = fmaxf(m, __shfl_xor(m, 8, 32));         // row max within 16-lane half
      float nm = fmaxf(rm[r], m);
      float alpha = __expf(rm[r] - nm);
      rm[r] = nm;
      float e0 = __expf(p0[r] - nm);
      float e1 = __expf(p1[r] - nm);
      p0[r] = e0; p1[r] = e1;
      float s = e0 + e1;
      s += __shfl_xor(s, 1, 32);
      s += __shfl_xor(s, 2, 32);
      s += __shfl_xor(s, 4, 32);
      s += __shfl_xor(s, 8, 32);
      rl[r] = rl[r] * alpha + s;
      o[0][r] *= alpha; o[1][r] *= alpha; o[2][r] *= alpha; o[3][r] *= alpha;
    }

    // transpose P (C layout -> A operand layout) through per-wave LDS tile.
    // LDS ops are in-order within a wave; no workgroup barrier needed.
#pragma unroll
    for (int r = 0; r < 8; ++r) {
      int row = k8 + r;
      pbuf[w][row][lr]      = (_Float16)p0[r];
      pbuf[w][row][16 + lr] = (_Float16)p1[r];
    }
    frag16 pf;
    pf.h[0] = *reinterpret_cast<const v8h*>(&pbuf[w][lr][k8]);
    pf.h[1] = *reinterpret_cast<const v8h*>(&pbuf[w][lr][k8 + 16]);

    v16h v0 = load_frag(&vbuf[cur][lr][k8]);
    v16h v1 = load_frag(&vbuf[cur][16 + lr][k8]);
    v16h v2 = load_frag(&vbuf[cur][32 + lr][k8]);
    v16h v3 = load_frag(&vbuf[cur][48 + lr][k8]);
    o[0] = wmma_f16(pf.v, v0, o[0]);
    o[1] = wmma_f16(pf.v, v1, o[1]);
    o[2] = wmma_f16(pf.v, v2, o[2]);
    o[3] = wmma_f16(pf.v, v3, o[3]);
  }

  if (active) {
    const float g = gates[0];
#pragma unroll
    for (int dn = 0; dn < 4; ++dn)
#pragma unroll
      for (int r = 0; r < 8; ++r) {
        int qrow = qt * 16 + k8 + r;
        if (qrow < QLEN)
          out[((size_t)b * QLEN + qrow) * EMB + h * HD + dn * 16 + lr] =
              o[dn][r] * (g / rl[r]);
      }
  }
}

// ---------------- launch ----------------

extern "C" void kernel_launch(void* const* d_in, const int* in_sizes, int n_in,
                              void* d_out, int out_size, void* d_ws, size_t ws_size,
                              hipStream_t stream) {
  (void)in_sizes; (void)n_in; (void)out_size; (void)ws_size;
  const float* x_b    = (const float*)d_in[1];
  const float* dbg    = (const float*)d_in[2];
  const float* Wk     = (const float*)d_in[3];
  const float* Wv     = (const float*)d_in[4];
  const float* Wpos   = (const float*)d_in[5];
  const float* prompt = (const float*)d_in[6];
  const float* gates  = (const float*)d_in[7];
  const float* pe     = (const float*)d_in[8];
  const int*   fb     = (const int*)d_in[9];    // full_b_step = 50
  const int*   fa     = (const int*)d_in[10];   // full_a_step = 801
  float* out = (float*)d_out;

  char* ws = (char*)d_ws;
  size_t off = 0;
  auto alloc = [&](size_t bytes) {
    void* p = ws + off;
    off = (off + bytes + 255) & ~(size_t)255;
    return p;
  };
  _Float16* kv16 = (_Float16*)alloc((size_t)BATCH * T_PAD * IN_DIM * 2);
  _Float16* k16  = (_Float16*)alloc((size_t)BATCH * T_PAD * EMB * 2);
  _Float16* vt16 = (_Float16*)alloc((size_t)BATCH * EMB * T_PAD * 2);
  _Float16* wk16 = (_Float16*)alloc((size_t)EMB * IN_DIM * 2);
  _Float16* wv16 = (_Float16*)alloc((size_t)EMB * IN_DIM * 2);
  _Float16* wp16 = (_Float16*)alloc((size_t)EMB * IN_DIM * 2);
  _Float16* posA = (_Float16*)alloc((size_t)QPAD * IN_DIM * 2);
  float*    posP = (float*)   alloc((size_t)QPAD * EMB * 4);
  _Float16* q16  = (_Float16*)alloc((size_t)BATCH * NH * QPAD * HD * 2);

  {
    size_t n = (size_t)BATCH * T_PAD * IN_DIM;
    build_kv<<<dim3((unsigned)((n + 255) / 256)), 256, 0, stream>>>(x_b, prompt, pe, kv16);
  }
  cvt_f16<<<dim3((EMB * IN_DIM + 255) / 256), 256, 0, stream>>>(Wk, wk16, EMB * IN_DIM);
  cvt_f16<<<dim3((EMB * IN_DIM + 255) / 256), 256, 0, stream>>>(Wv, wv16, EMB * IN_DIM);
  cvt_f16<<<dim3((EMB * IN_DIM + 255) / 256), 256, 0, stream>>>(Wpos, wp16, EMB * IN_DIM);
  build_posA<<<dim3((QPAD * IN_DIM + 255) / 256), 256, 0, stream>>>(pe, fb, fa, posA);

  // 16 M-tiles (256 rows) per block; T_PAD/16 = 130 tiles -> grid.y = 9
  gemm16<<<dim3(EMB / 64, 9, BATCH), 256, 0, stream>>>(kv16, wk16, (void*)k16, T_PAD / 16, 0);
  gemm16<<<dim3(EMB / 64, 9, BATCH), 256, 0, stream>>>(kv16, wv16, (void*)vt16, T_PAD / 16, 1);
  gemm16<<<dim3(EMB / 64, 4, 1), 256, 0, stream>>>(posA, wp16, (void*)posP, QPAD / 16, 2);

  {
    size_t n = (size_t)BATCH * NH * QPAD * HD;
    build_q<<<dim3((unsigned)((n + 255) / 256)), 256, 0, stream>>>(dbg, posP, q16);
  }
  attn<<<dim3((QTILES + 7) / 8, NH, BATCH), 256, 0, stream>>>(q16, k16, vt16, gates, out);
}
